// AdvancedChimeraLayer_9234179686589
// MI455X (gfx1250) — matrix operand order repaired and checked by reading, not compiled, not run on hardware
//
#include <hip/hip_runtime.h>
#include <hip/hip_bf16.h>
#include <math.h>

// ---- problem constants (from reference setup_inputs) ----
constexpr int B_DIM = 4;
constexpr int S_DIM = 8192;
constexpr int D_DIM = 2048;
constexpr int BUS   = 128;
constexpr int LCCH  = 24;      // cached layers
constexpr int LAUG  = 25;      // L+1
constexpr int TILE_M = 128;    // rows per workgroup in fused kernel

typedef __bf16 v16bf __attribute__((ext_vector_type(16)));
typedef float  v8f   __attribute__((ext_vector_type(8)));

__device__ __forceinline__ __bf16 f2bf(float f) { return (__bf16)f; }

// pack two fp32 -> packed bf16 dword (lowers to v_cvt_pk_bf16_f32)
__device__ __forceinline__ unsigned pk2(float a, float b) {
  unsigned lo = __builtin_bit_cast(unsigned short, (__bf16)a);
  unsigned hi = __builtin_bit_cast(unsigned short, (__bf16)b);
  return lo | (hi << 16);
}

__device__ __forceinline__ v8f zero8() {
  v8f c;
#pragma unroll
  for (int i = 0; i < 8; ++i) c[i] = 0.0f;
  return c;
}

// convert fp32 -> v16bf (B-matrix fragment: elements 0..7 = p[0..7], 8..15 = p[16..23])
__device__ __forceinline__ v16bf cvt16(const float* __restrict__ p) {
  v16bf r;
#pragma unroll
  for (int i = 0; i < 16; ++i) r[i] = f2bf(p[(i < 8) ? i : i + 8]);
  return r;
}

// A-matrix fragment from LDS bf16: elements 0..7 = p[0..7], 8..15 = p[16..23]
__device__ __forceinline__ v16bf afrag_lds(const __bf16* p) {
  v16bf r;
#pragma unroll
  for (int i = 0; i < 8; ++i) r[i] = p[i];
#pragma unroll
  for (int i = 0; i < 8; ++i) r[8 + i] = p[16 + i];
  return r;
}

// ------------------------------------------------------------------
// Kernel 1: g[b,s] = sigmoid(dot(x[b,s,:], W_pool_gate))   (wave per row)
// ------------------------------------------------------------------
__global__ __launch_bounds__(256)
void pool_gate_kernel(const float* __restrict__ x, const float* __restrict__ Wg,
                      float* __restrict__ g) {
  int wave = threadIdx.x >> 5;
  int lane = threadIdx.x & 31;
  size_t row = (size_t)blockIdx.x * 8 + wave;           // row in [0, B*S)
  const float* xr = x + row * D_DIM;
  float s = 0.0f;
#pragma unroll 4
  for (int k = lane * 4; k < D_DIM; k += 128) {
    float4 v = *(const float4*)(xr + k);
    float4 w = *(const float4*)(Wg + k);
    s += v.x * w.x + v.y * w.y + v.z * w.z + v.w * w.w;
  }
#pragma unroll
  for (int off = 16; off > 0; off >>= 1) s += __shfl_xor(s, off, 32);
  if (lane == 0) g[row] = 1.0f / (1.0f + expf(-s));
}

// ------------------------------------------------------------------
// Kernel 2: xp[b,d] = sum_s x[b,s,d] * g[b,s]    (column-parallel)
// ------------------------------------------------------------------
__global__ __launch_bounds__(256)
void pool_sum_kernel(const float* __restrict__ x, const float* __restrict__ g,
                     float* __restrict__ xp) {
  __shared__ float gch[256];
  int b = blockIdx.y;
  int d = blockIdx.x * 256 + threadIdx.x;
  const float* xb = x + (size_t)b * S_DIM * D_DIM;
  const float* gb = g + (size_t)b * S_DIM;
  float acc = 0.0f;
  for (int s0 = 0; s0 < S_DIM; s0 += 256) {
    __syncthreads();
    gch[threadIdx.x] = gb[s0 + threadIdx.x];
    __syncthreads();
#pragma unroll 8
    for (int j = 0; j < 256; ++j)
      acc += xb[(size_t)(s0 + j) * D_DIM + d] * gch[j];
  }
  xp[b * D_DIM + d] = acc;
}

// ------------------------------------------------------------------
// Kernel 3: summary[b] = l2norm((xp/(sum_g+eps)) @ W_publish^T); emits new_cache
// ------------------------------------------------------------------
__global__ __launch_bounds__(128)
void summary_kernel(const float* __restrict__ g, const float* __restrict__ xp,
                    const float* __restrict__ Wpub, const float* __restrict__ bus_cache,
                    float* __restrict__ summary, float* __restrict__ out_cache) {
  __shared__ float red[128];
  int b = blockIdx.x, t = threadIdx.x;
  // denominator
  float sg = 0.0f;
  for (int s = t; s < S_DIM; s += 128) sg += g[(size_t)b * S_DIM + s];
  red[t] = sg;
  __syncthreads();
  for (int off = 64; off > 0; off >>= 1) {
    if (t < off) red[t] += red[t + off];
    __syncthreads();
  }
  float denom = red[0] + 1e-6f;
  __syncthreads();
  // s_o = dot(xp, Wpub[o]) / denom
  const float* xpb = xp + b * D_DIM;
  const float* wr  = Wpub + (size_t)t * D_DIM;
  float acc = 0.0f;
#pragma unroll 4
  for (int k = 0; k < D_DIM; ++k) acc += xpb[k] * wr[k];
  acc /= denom;
  // l2 norm over 128 values
  red[t] = acc * acc;
  __syncthreads();
  for (int off = 64; off > 0; off >>= 1) {
    if (t < off) red[t] += red[t + off];
    __syncthreads();
  }
  float nrm = sqrtf(red[0]);
  float val = acc / fmaxf(nrm, 1e-12f);
  summary[b * BUS + t] = val;
  // new_cache = concat(bus_cache, summary)
  float* oc = out_cache + (size_t)b * LAUG * BUS;
  for (int l = 0; l < LCCH; ++l)
    oc[l * BUS + t] = bus_cache[((size_t)b * LCCH + l) * BUS + t];
  oc[LCCH * BUS + t] = val;
}

// ------------------------------------------------------------------
// Kernel 4 (fused): q GEMM (WMMA bf16, double-buffered LDS staging) ->
//                   softmax over 25 -> gather -> modulation GEMM (WMMA bf16)
//                   + residual add.  grid = (S/128, B), 256 threads (8 waves).
// ------------------------------------------------------------------
__global__ __launch_bounds__(256)
void fused_attn_kernel(const float* __restrict__ x, const float* __restrict__ bus_cache,
                       const float* __restrict__ Wq, const float* __restrict__ Wmod,
                       const float* __restrict__ summary, const float* __restrict__ gate,
                       float* __restrict__ xout) {
  __shared__ __bf16 qg[TILE_M][BUS + 2];     // q (phase A/B) then gathered (phase C)
  __shared__ float  aug[LAUG][BUS];          // keys/values (fp32 for softmax precision)
  __shared__ __bf16 xch[2][TILE_M][34];      // double-buffered bf16 K-chunks of x
  // total LDS: 33,280 + 12,800 + 17,408 = 63,488 B

  const int tid   = threadIdx.x;
  const int wave  = tid >> 5;
  const int lane  = tid & 31;
  const int lhalf = lane & 15;
  const int lsel  = lane >> 4;               // 0: lanes 0-15, 1: lanes 16-31
  const int b     = blockIdx.y;
  const int mrow0 = blockIdx.x * TILE_M;

  const float* xb = x    + ((size_t)b * S_DIM + mrow0) * D_DIM;
  float*       ob = xout + ((size_t)b * S_DIM + mrow0) * D_DIM;

  // packed bf16 staging of one 128x32 x-chunk (8 dword stores per thread)
  auto stage = [&](int bsel, int kk) {
    const int r  = tid >> 1;
    const int c0 = (tid & 1) * 16;
    const float* src = xb + (size_t)r * D_DIM + kk + c0;
    float4 f0 = *(const float4*)(src);
    float4 f1 = *(const float4*)(src + 4);
    float4 f2 = *(const float4*)(src + 8);
    float4 f3 = *(const float4*)(src + 12);
    unsigned* dst = (unsigned*)&xch[bsel][r][c0];   // 4B-aligned
    dst[0] = pk2(f0.x, f0.y); dst[1] = pk2(f0.z, f0.w);
    dst[2] = pk2(f1.x, f1.y); dst[3] = pk2(f1.z, f1.w);
    dst[4] = pk2(f2.x, f2.y); dst[5] = pk2(f2.z, f2.w);
    dst[6] = pk2(f3.x, f3.y); dst[7] = pk2(f3.z, f3.w);
  };

  // ---------------- Phase A: q = x_tile @ Wq^T  (wave w owns N cols 16w..16w+15)
  v8f acc[8];
#pragma unroll
  for (int i = 0; i < 8; ++i) acc[i] = zero8();

  const int nn = wave * 16;
  const float* wqrow = Wq + (size_t)(nn + lhalf) * D_DIM;   // B[k][n] = Wq[nn+n][k]

  stage(0, 0);
  // hoisted cooperative load of aug = [summary; bus_cache] (overlaps GEMM1)
  for (int idx = tid; idx < LAUG * BUS; idx += 256) {
    int l = idx >> 7, o = idx & 127;
    aug[l][o] = (l == 0) ? summary[b * BUS + o]
                         : bus_cache[((size_t)b * LCCH + (l - 1)) * BUS + o];
  }
  __syncthreads();

  int buf = 0;
  for (int kk = 0; kk < D_DIM; kk += 32) {
    if (kk + 32 < D_DIM) stage(buf ^ 1, kk + 32);  // prefetch next chunk into other buffer
    v16bf bfr = cvt16(wqrow + kk + lsel * 8);      // K = kk+{0..7,16..23} (lo) / kk+{8..15,24..31} (hi)
#pragma unroll
    for (int mt = 0; mt < 8; ++mt) {
      v16bf afr = afrag_lds(&xch[buf][mt * 16 + lhalf][lsel * 8]);
      acc[mt] = __builtin_amdgcn_wmma_f32_16x16x32_bf16(
          false, afr, false, bfr, (short)0, acc[mt], false, false);
    }
    __syncthreads();   // single barrier per iteration (double buffering)
    buf ^= 1;
  }

  // write q (bf16) to LDS; C layout: VGPR i -> row mt*16 + i + 8*lsel, col nn+lhalf
#pragma unroll
  for (int mt = 0; mt < 8; ++mt) {
    int rb = mt * 16 + lsel * 8;
#pragma unroll
    for (int i = 0; i < 8; ++i) qg[rb + i][nn + lhalf] = f2bf(acc[mt][i]);
  }
  __syncthreads();

  // ---------------- Phase B: scores -> softmax(25) -> gathered
  // 2 threads per row (all 256 threads busy); partials combined via shfl_xor(1).
  const int prow = tid >> 1;
  const int o0   = (tid & 1) * 64;
  float attnv[LAUG];
  {
    float sc[LAUG];
#pragma unroll
    for (int l = 0; l < LAUG; ++l) sc[l] = 0.0f;
    for (int o = o0; o < o0 + 64; ++o) {
      float qv = (float)qg[prow][o];
#pragma unroll
      for (int l = 0; l < LAUG; ++l) sc[l] += qv * aug[l][o];
    }
#pragma unroll
    for (int l = 0; l < LAUG; ++l) sc[l] += __shfl_xor(sc[l], 1, 32);
    const float scale = 0.0883883476483184f;  // 1/sqrt(128)
    float mx = -INFINITY;
#pragma unroll
    for (int l = 0; l < LAUG; ++l) { sc[l] *= scale; mx = fmaxf(mx, sc[l]); }
    float se = 0.0f;
#pragma unroll
    for (int l = 0; l < LAUG; ++l) { sc[l] = expf(sc[l] - mx); se += sc[l]; }
    float inv = 1.0f / se;
#pragma unroll
    for (int l = 0; l < LAUG; ++l) attnv[l] = sc[l] * inv;
  }
  __syncthreads();   // all q reads done before overwriting with gathered
  for (int o = o0; o < o0 + 64; o += 2) {
    float g0 = 0.0f, g1 = 0.0f;
#pragma unroll
    for (int l = 0; l < LAUG; ++l) {
      g0 += attnv[l] * aug[l][o];
      g1 += attnv[l] * aug[l][o + 1];
    }
    *(unsigned*)&qg[prow][o] = pk2(g0, g1);   // packed bf16 store
  }
  __syncthreads();

  // ---------------- Phase C: modulation = gathered @ Wmod^T, + residual
  // wave w owns d columns [w*256, w*256+256); Wmod read once per WG (L2-resident).
  const float sgate = 1.0f / (1.0f + expf(-gate[0]));
  const int ddbase = wave * 256;
  for (int dt = 0; dt < 256; dt += 16) {
    const int dd = ddbase + dt;
    v16bf bfr[4];
    const float* wm = Wmod + (size_t)(dd + lhalf) * BUS + lsel * 8;
#pragma unroll
    for (int ot = 0; ot < 4; ++ot) bfr[ot] = cvt16(wm + ot * 32);

#pragma unroll 2
    for (int mt = 0; mt < 8; ++mt) {
      const int mbase = mt * 16;
      v8f c = zero8();
#pragma unroll
      for (int ot = 0; ot < 4; ++ot) {
        v16bf afr = afrag_lds(&qg[mbase + lhalf][ot * 32 + lsel * 8]);
        c = __builtin_amdgcn_wmma_f32_16x16x32_bf16(
            false, afr, false, bfr[ot], (short)0, c, false, false);
      }
      // fused epilogue: out = x + sigmoid(gate) * modulation
      const int col = dd + lhalf;
      const int r0  = mbase + lsel * 8;
#pragma unroll
      for (int i = 0; i < 8; ++i) {
        size_t idx = (size_t)(r0 + i) * D_DIM + col;
        ob[idx] = xb[idx] + c[i] * sgate;
      }
    }
  }
}

// ------------------------------------------------------------------
extern "C" void kernel_launch(void* const* d_in, const int* in_sizes, int n_in,
                              void* d_out, int out_size, void* d_ws, size_t ws_size,
                              hipStream_t stream) {
  (void)in_sizes; (void)n_in; (void)out_size; (void)ws_size;
  const float* x    = (const float*)d_in[0];
  const float* bus  = (const float*)d_in[1];
  const float* Wpub = (const float*)d_in[2];
  const float* Wq   = (const float*)d_in[3];
  const float* Wmod = (const float*)d_in[4];
  const float* Wg   = (const float*)d_in[5];
  const float* gate = (const float*)d_in[6];

  float* xout   = (float*)d_out;
  float* ocache = xout + (size_t)B_DIM * S_DIM * D_DIM;   // new_cache after x_out

  float* g    = (float*)d_ws;                 // [B*S]
  float* xp   = g + (size_t)B_DIM * S_DIM;    // [B*D]
  float* summ = xp + (size_t)B_DIM * D_DIM;   // [B*BUS]

  pool_gate_kernel<<<(B_DIM * S_DIM) / 8, 256, 0, stream>>>(x, Wg, g);
  pool_sum_kernel<<<dim3(D_DIM / 256, B_DIM), 256, 0, stream>>>(x, g, xp);
  summary_kernel<<<B_DIM, 128, 0, stream>>>(g, xp, Wpub, bus, summ, ocache);
  fused_attn_kernel<<<dim3(S_DIM / TILE_M, B_DIM), 256, 0, stream>>>(
      x, bus, Wq, Wmod, summ, gate, xout);
}
